// GenerativeGNN_22574348108073
// MI455X (gfx1250) — matrix operand order repaired and checked by reading, not compiled, not run on hardware
//
#include <hip/hip_runtime.h>

// ---------------------------------------------------------------------------
// GCN (3 layers) on MI455X / gfx1250.
//   N=50000 nodes, E=800000 edges, C=128 channels, DE=16 edge-attr dims.
// - Working set (H, AGG, edge_attr, edge_index) is L2-resident (192 MB L2);
//   ea = edge_attr @ emb_ea (410 MB if materialized) is recomputed per edge
//   from LDS-resident emb_ea instead.
// - Dense GEMMs: split-bf16 WMMA (hi/lo decomposition, f32 accumulate) for
//   ~f32 accuracy on the fast bf16 matrix path.
// - A tile is async-staged (GLOBAL_LOAD_ASYNC_TO_LDS_B128, ASYNCcnt) then
//   converted ONCE cooperatively into fragment-ordered bf16 hi/lo LDS planes
//   (no per-wave conversion redundancy); WMMA loop is pure ds_load + wmma.
// - B (weight) fragments are PRE-PACKED into the WMMA lane layout so each
//   lane loads its fragment as one contiguous 32B chunk.
// - Scatter uses hardware global_atomic_add_f32 (L2-resident).
// ---------------------------------------------------------------------------

#define NN   50000
#define EE   800000
#define CC   128
#define DEE  16

typedef __attribute__((ext_vector_type(16))) __bf16 v16bf;
typedef __attribute__((ext_vector_type(8)))  float  v8f;

// Async global->LDS copy of 16 bytes per lane (CDNA5 async path, ASYNCcnt).
// lds_off: LDS byte offset (flat shared address truncated to 32 bits, per
// ISA aperture rule: the low 32 bits are the LDS offset).  gptr: 64-bit global address.
__device__ __forceinline__ void async_copy_b128(unsigned lds_off,
                                                const void* gptr) {
  asm volatile("global_load_async_to_lds_b128 %0, %1, off"
               :: "v"(lds_off), "v"(gptr)
               : "memory");
}
__device__ __forceinline__ void wait_async0() {
  asm volatile("s_wait_asynccnt 0" ::: "memory");
}

__device__ __forceinline__ unsigned short bf16_bits(__bf16 h) {
  union { __bf16 b; unsigned short u; } c; c.b = h; return c.u;
}

// ---------------------------------------------------------------------------
// Pack W1..W3 (f32 128x128) into bf16 hi/lo planes laid out EXACTLY as the
// WMMA B fragments: Bpack[mat][kt][nt][lane][j], j=0..15 contiguous.
//   B (16-bit, 32x16 KxN, contiguous k order used here, same as A): lane L: N = nt*16 + (L%16),
//   half = L/16, element j -> K = kt*32 + half*16 + j.
// One thread per packed element (bijective permutation of W's elements).
// ---------------------------------------------------------------------------
__global__ void gnn_pack_weights(const float* __restrict__ W1,
                                 const float* __restrict__ W2,
                                 const float* __restrict__ W3,
                                 __bf16* __restrict__ Bhi,
                                 __bf16* __restrict__ Blo) {
  int i = blockIdx.x * blockDim.x + threadIdx.x;   // 0 .. 3*16384-1
  int mat = i >> 14;
  int off = i & 16383;
  int j    = off & 15;
  int lane = (off >> 4) & 31;
  int nt   = (off >> 9) & 7;
  int kt   = (off >> 12) & 3;

  int kB = kt * 32 + (lane >> 4) * 16 + j;
  int n  = nt * 16 + (lane & 15);

  const float* W = (mat == 0) ? W1 : (mat == 1) ? W2 : W3;
  float v = W[kB * CC + n];
  __bf16 h = (__bf16)v;
  Bhi[i] = h;
  Blo[i] = (__bf16)(v - (float)h);
}

// ---------------------------------------------------------------------------
// Degree: zero, atomic-count on col, then deg^-0.5 (0 for isolated nodes).
// ---------------------------------------------------------------------------
__global__ void gnn_zero_deg(float* __restrict__ deg) {
  int i = blockIdx.x * blockDim.x + threadIdx.x;
  if (i < NN) deg[i] = 0.0f;
}

__global__ void gnn_count_deg(const int* __restrict__ eidx,
                              float* __restrict__ deg) {
  int e = blockIdx.x * blockDim.x + threadIdx.x;
  if (e < EE) {
    int c = eidx[EE + e];                 // col = edge_index[1][e]
    unsafeAtomicAdd(&deg[c], 1.0f);       // global_atomic_add_f32 (no-return)
  }
}

__global__ void gnn_inv_sqrt_deg(const float* __restrict__ deg,
                                 float* __restrict__ dis) {
  int i = blockIdx.x * blockDim.x + threadIdx.x;
  if (i < NN) {
    float d = deg[i];
    dis[i] = (d > 0.0f) ? rsqrtf(d) : 0.0f;
  }
}

// ---------------------------------------------------------------------------
// H = relu?(IN) @ W + b ; also AGG = H (residual init for the scatter pass).
//   Block: 256 threads = 8 waves. One 16-row M-tile per block; each wave
//   owns one 16-col N-tile. K-loop: 4 steps of K=32.
//   Split-bf16: C += Ahi*Bhi + Alo*Bhi + Ahi*Blo  (3 WMMAs / k-step).
// A fragment (16-bit A 16x32, contiguous k order used here): lane L: M=L%16, half=L/16,
//   elem j -> K = half*16 + j.
// Fragment-ordered LDS planes: AhiL/AloL[(kt*2+half)*16+mrow][j] contiguous,
// so a fragment is one 32B LDS chunk per lane (2x ds_load_b128).
// C/D (f32 16x16): vgpr r: M = half*8 + r, N = L%16.
// ---------------------------------------------------------------------------
__global__ void gnn_gemm_wmma(const float*  __restrict__ IN,
                              const __bf16* __restrict__ Bhi,
                              const __bf16* __restrict__ Blo,
                              const float*  __restrict__ bias,
                              float* __restrict__ H,
                              float* __restrict__ AGG,
                              int do_relu) {
  __shared__ float  Atile[16 * CC];                // 8 KB: f32 staging
  __shared__ __bf16 AhiL[16 * CC];                 // 4 KB: fragment-ordered hi
  __shared__ __bf16 AloL[16 * CC];                 // 4 KB: fragment-ordered lo

  const int tid   = threadIdx.x;
  const int mbase = blockIdx.x * 16;               // 3125 blocks * 16 = 50000

  // Async-stage the 16x128 f32 tile (contiguous 8 KB) into LDS.
  {
    const char* src = (const char*)(IN + (size_t)mbase * CC);
    unsigned lds0 = (unsigned)(uintptr_t)(void*)Atile;   // addr[31:0] = LDS off
    for (int i = tid; i < (16 * CC * 4) / 16; i += 256)  // 512 x 16B
      async_copy_b128(lds0 + i * 16, src + i * 16);
    wait_async0();
  }
  __syncthreads();

  const float relu_lo = do_relu ? 0.0f : -__builtin_inff();

  // One cooperative conversion pass: 2048 elements, 2 per iteration so the
  // bf16 hi/lo pairs pack into single b32 LDS stores.
  {
    unsigned* AhiP = (unsigned*)AhiL;
    unsigned* AloP = (unsigned*)AloL;
    for (int p = tid; p < (16 * CC) / 2; p += 256) {     // 1024 pairs
      int j0   = (p << 1) & 15;                          // even j
      int mrow = (p >> 3) & 15;
      int half = (p >> 7) & 1;
      int kt   = (p >> 8) & 3;
      int kA0 = kt * 32 + half * 16 + j0;
      float a0 = fmaxf(Atile[mrow * CC + kA0],     relu_lo);  // fused ReLU
      float a1 = fmaxf(Atile[mrow * CC + kA0 + 1], relu_lo);  // j0&7 < 7
      __bf16 h0 = (__bf16)a0, h1 = (__bf16)a1;
      float  l0 = a0 - (float)h0, l1 = a1 - (float)h1;
      AhiP[p] = (unsigned)bf16_bits(h0) | ((unsigned)bf16_bits(h1) << 16);
      AloP[p] = (unsigned)bf16_bits((__bf16)l0) |
                ((unsigned)bf16_bits((__bf16)l1) << 16);
    }
  }
  __syncthreads();

  const int wave = tid >> 5;                       // N-tile id, 0..7
  const int lane = tid & 31;
  const int half = lane >> 4;
  const int mrow = lane & 15;
  const int ncol = wave * 16 + (lane & 15);

  v8f acc = {};

#pragma unroll
  for (int kt = 0; kt < 4; ++kt) {
    // B fragments: one contiguous 32B chunk per lane, coalesced per wave.
    const size_t boff = (size_t)(((kt * 8 + wave) * 32 + lane) * 16);
    const v16bf bhi = *(const v16bf*)(Bhi + boff);
    const v16bf blo = *(const v16bf*)(Blo + boff);

    // A fragments: one contiguous 32B LDS chunk per lane.
    const int afrag = ((kt * 2 + half) * 16 + mrow) * 16;
    const v16bf ahi = *(const v16bf*)&AhiL[afrag];
    const v16bf alo = *(const v16bf*)&AloL[afrag];

    // D = A*B + C, f32 accumulate (args: neg_a, A, neg_b, B, c_mod, C, ra, rb)
    acc = __builtin_amdgcn_wmma_f32_16x16x32_bf16(false, ahi, false, bhi,
                                                  (short)0, acc, false, false);
    acc = __builtin_amdgcn_wmma_f32_16x16x32_bf16(false, alo, false, bhi,
                                                  (short)0, acc, false, false);
    acc = __builtin_amdgcn_wmma_f32_16x16x32_bf16(false, ahi, false, blo,
                                                  (short)0, acc, false, false);
  }

  const float bv = bias[ncol];
#pragma unroll
  for (int r = 0; r < 8; ++r) {
    int row = mbase + half * 8 + r;
    float v = acc[r] + bv;
    size_t idx = (size_t)row * CC + ncol;
    H[idx]   = v;      // pristine h for the gather
    AGG[idx] = v;      // residual: AGG starts at h, scatter adds on top
  }
}

// ---------------------------------------------------------------------------
// Edge pass: one wave32 per edge, 4 channels per lane.
//   norm = dis[row]*dis[col]
//   msg  = norm * (H[row] + edge_attr[e] @ emb_ea)   (emb_ea async-staged)
//   AGG[col] += msg   via hardware f32 atomics (L2-resident)
// ---------------------------------------------------------------------------
__global__ void gnn_edge_scatter(const int*   __restrict__ eidx,
                                 const float* __restrict__ eattr,
                                 const float* __restrict__ emb,   // [16][128]
                                 const float* __restrict__ dis,
                                 const float* __restrict__ H,
                                 float* __restrict__ AGG) {
  __shared__ float semb[DEE * CC];                 // 8 KB
  const int tid = threadIdx.x;
  {
    unsigned lds0 = (unsigned)(uintptr_t)(void*)semb;
    const char* src = (const char*)emb;
    for (int i = tid; i < (DEE * CC * 4) / 16; i += 256)   // 512 x 16B
      async_copy_b128(lds0 + i * 16, src + i * 16);
    wait_async0();
  }
  __syncthreads();

  const int e    = blockIdx.x * 8 + (tid >> 5);    // 100000 blocks * 8 waves
  const int lane = tid & 31;

  __builtin_prefetch(&eattr[(size_t)e * DEE], 0, 0);   // global_prefetch_b8

  const int r = eidx[e];
  const int c = eidx[EE + e];
  const float nrm = dis[r] * dis[c];

  const int ch = lane * 4;
  const float4 h4 = *(const float4*)&H[(size_t)r * CC + ch];

  float ax = 0.0f, ay = 0.0f, az = 0.0f, aw = 0.0f;
#pragma unroll
  for (int k = 0; k < DEE; ++k) {
    float a = eattr[(size_t)e * DEE + k];          // wave-uniform row
    const float4 em = *(const float4*)&semb[k * CC + ch];
    ax = fmaf(a, em.x, ax);
    ay = fmaf(a, em.y, ay);
    az = fmaf(a, em.z, az);
    aw = fmaf(a, em.w, aw);
  }

  float* dst = &AGG[(size_t)c * CC + ch];
  unsafeAtomicAdd(dst + 0, nrm * (h4.x + ax));
  unsafeAtomicAdd(dst + 1, nrm * (h4.y + ay));
  unsafeAtomicAdd(dst + 2, nrm * (h4.z + az));
  unsafeAtomicAdd(dst + 3, nrm * (h4.w + aw));
}

// ---------------------------------------------------------------------------
// Launch
// ---------------------------------------------------------------------------
extern "C" void kernel_launch(void* const* d_in, const int* in_sizes, int n_in,
                              void* d_out, int out_size, void* d_ws, size_t ws_size,
                              hipStream_t stream) {
  (void)in_sizes; (void)n_in; (void)out_size; (void)ws_size;

  const float* x     = (const float*)d_in[0];
  const int*   eidx  = (const int*)  d_in[1];
  const float* eattr = (const float*)d_in[2];
  const float* emb   = (const float*)d_in[3];
  const float* W1    = (const float*)d_in[4];
  const float* b1    = (const float*)d_in[5];
  const float* W2    = (const float*)d_in[6];
  const float* b2    = (const float*)d_in[7];
  const float* W3    = (const float*)d_in[8];
  const float* b3    = (const float*)d_in[9];
  float* out = (float*)d_out;

  // Workspace carving (all offsets 1 KB aligned).
  char* ws = (char*)d_ws;
  size_t off = 0;
  auto take = [&](size_t bytes) {
    void* p = ws + off;
    off += (bytes + 1023) & ~(size_t)1023;
    return p;
  };
  float*  deg  = (float*) take((size_t)NN * 4);
  float*  dis  = (float*) take((size_t)NN * 4);
  __bf16* Bhi  = (__bf16*)take((size_t)3 * CC * CC * 2);
  __bf16* Blo  = (__bf16*)take((size_t)3 * CC * CC * 2);
  float*  Hbuf = (float*) take((size_t)NN * CC * 4);
  float*  A0   = (float*) take((size_t)NN * CC * 4);
  float*  A1   = (float*) take((size_t)NN * CC * 4);

  // --- prep: weight pack, degree normalization (shared by all 3 layers) ---
  gnn_pack_weights<<<(3 * CC * CC) / 256, 256, 0, stream>>>(W1, W2, W3, Bhi, Blo);
  gnn_zero_deg<<<(NN + 255) / 256, 256, 0, stream>>>(deg);
  gnn_count_deg<<<(EE + 255) / 256, 256, 0, stream>>>(eidx, deg);
  gnn_inv_sqrt_deg<<<(NN + 255) / 256, 256, 0, stream>>>(deg, dis);

  const int gemm_grid = NN / 16;   // 3125
  const int edge_grid = EE / 8;    // 100000

  // --- layer 1: in = x (no relu) -> A0 ---
  gnn_gemm_wmma<<<gemm_grid, 256, 0, stream>>>(x, Bhi, Blo, b1, Hbuf, A0, 0);
  gnn_edge_scatter<<<edge_grid, 256, 0, stream>>>(eidx, eattr, emb, dis, Hbuf, A0);

  // --- layer 2: in = relu(A0) -> A1 ---
  gnn_gemm_wmma<<<gemm_grid, 256, 0, stream>>>(A0, Bhi + 16384, Blo + 16384, b2,
                                               Hbuf, A1, 1);
  gnn_edge_scatter<<<edge_grid, 256, 0, stream>>>(eidx, eattr, emb, dis, Hbuf, A1);

  // --- layer 3: in = relu(A1) -> d_out ---
  gnn_gemm_wmma<<<gemm_grid, 256, 0, stream>>>(A1, Bhi + 32768, Blo + 32768, b3,
                                               Hbuf, out, 1);
  gnn_edge_scatter<<<edge_grid, 256, 0, stream>>>(eidx, eattr, emb, dis, Hbuf, out);
}
